// SAM3A_27934467293447
// MI455X (gfx1250) — matrix operand order repaired and checked by reading, not compiled, not run on hardware
//
#include <hip/hip_runtime.h>
#include <hip/hip_bf16.h>

#define BB 8192
#define NF 64
#define DD 128
#define GB 16       // batches per workgroup
#define ROWP 136    // padded LDS row for F/P tiles (bf16 elems): 272 B, 16B aligned
#define SROW 4104   // per-batch S block stride (bf16 elems): 8208 B, 16B aligned,
                    // 2052 dwords == 4 (mod 64) -> conflict-free strided A loads

#if __has_builtin(__builtin_amdgcn_global_load_async_to_lds_b128)
#define HAVE_ASYNC 1
#else
#define HAVE_ASYNC 0
#endif

#if __has_builtin(__builtin_amdgcn_s_wait_asynccnt)
#define WAITA(N) __builtin_amdgcn_s_wait_asynccnt(N)
#else
#define WAITA(N) asm volatile("s_wait_asynccnt " #N ::: "memory")
#endif

typedef __attribute__((ext_vector_type(4)))  int    v4i;
typedef __attribute__((address_space(1))) v4i*       gv4i_p;   // global src (non-const!)
typedef __attribute__((address_space(3))) v4i*       lv4i_p;   // LDS dst

typedef __attribute__((ext_vector_type(16))) __bf16 v16bf;
typedef __attribute__((ext_vector_type(8)))  __bf16 v8bf;
typedef __attribute__((ext_vector_type(8)))  float  v8f;

union V16U { v16bf v; v8bf h[2]; };

// A fragment (16x32 bf16, M x K) from row-major storage, row stride ld (elems).
// Lane: row = lane&15; kb = (lane>>4)*8; elems 0..7 = K[kb..kb+7],
// elems 8..15 = K[kb+16..kb+23].  (ISA 7.12.2, 16-bit A 16x32)
__device__ inline v16bf ld_a(const __bf16* base, int ld, int lane) {
  const int row = lane & 15;
  const int kb  = (lane >> 4) << 3;
  const __bf16* p = base + row * ld + kb;
  V16U u;
  u.h[0] = *(const v8bf*)(p);
  u.h[1] = *(const v8bf*)(p + 16);
  return u.v;
}

// B fragment (32x16 bf16, K x N) from N-major (transposed) storage: caller
// passes p = &BT[n][k0 + kbase], n = lane&15, kbase = (lane>>4)*8.
__device__ inline v16bf ld_b(const __bf16* p) {
  V16U u;
  u.h[0] = *(const v8bf*)(p);
  u.h[1] = *(const v8bf*)(p + 16);
  return u.v;
}

__device__ inline v8f wmma_bf16(v16bf a, v16bf b, v8f c) {
  return __builtin_amdgcn_wmma_f32_16x16x32_bf16(
      /*neg_a=*/false, a, /*neg_b=*/false, b,
      /*c_mod=*/(short)0, c, /*reuse_a=*/false, /*reuse_b=*/false);
}

// ---------------------------------------------------------------------------
// Kernel 0: bf16 conversions.  K_w/Q_w row-major == N-major operand for
// B = K_w^T / Q_w^T.  W[i][j][d] -> Wt[i][d][j] (N-major operand for step 3).
// ---------------------------------------------------------------------------
__global__ __launch_bounds__(256) void k_convert(
    const float* __restrict__ Kw, const float* __restrict__ Qw,
    const float* __restrict__ W,
    __bf16* __restrict__ Kwbf, __bf16* __restrict__ Qwbf,
    __bf16* __restrict__ Wt) {
  const int idx = blockIdx.x * 256 + threadIdx.x;
  if (idx < DD * DD) {
    Kwbf[idx] = (__bf16)Kw[idx];
    Qwbf[idx] = (__bf16)Qw[idx];
  }
  const int total = NF * NF * DD;
  for (int t = idx; t < total; t += gridDim.x * 256) {
    const int i = t / (NF * DD);
    const int r = t % (NF * DD);
    const int d = r / NF;
    const int j = r % NF;
    Wt[t] = (__bf16)W[(i * NF + j) * DD + d];
  }
}

#if HAVE_ASYNC
// Issue async copy of one batch's F slab (32 KB) into LDS raw buffer.
// 256 threads x 8 x 16B, ASYNCcnt += 8 per wave, completes in order.
__device__ inline void stage_async(const float* Fb, float* sraw, int tid) {
#pragma unroll
  for (int it = 0; it < 8; ++it) {
    const int idx = (tid + it * 256) * 4;   // float index, 16B granules
    __builtin_amdgcn_global_load_async_to_lds_b128(
        (gv4i_p)(Fb + idx), (lv4i_p)(sraw + idx), 0, 0);
  }
}
#endif

// ---------------------------------------------------------------------------
// Fused kernel: 1 WG = GB batches, 8 waves.
// Phase A (per batch): async-stage F_b -> LDS (double-buffered) -> bf16;
//   P = F@Kw^T -> LDS and residual F@Qw^T -> Out in ONE pass (shared A frags);
//   S = F@P^T -> LDS (never leaves the WGP).
// Phase B: for each i: Out[b0..b0+15, i, :] += S_tile @ W[i]  (RMW on this
//   WG's freshly written Out lines -> absorbed by L2).
// ---------------------------------------------------------------------------
__global__ __launch_bounds__(256) void k_fused(
    const float* __restrict__ F,
    const __bf16* __restrict__ Kwbf, const __bf16* __restrict__ Qwbf,
    const __bf16* __restrict__ Wt,
    float* __restrict__ Out) {
  __shared__ __bf16 sF[NF * ROWP];   // current batch F (bf16)
  __shared__ __bf16 sP[NF * ROWP];   // P = F@Kw^T (row-major == N-major B^T)
  __shared__ __bf16 sS[GB * SROW];   // S for all GB batches: [m][i*64 + j]
#if HAVE_ASYNC
  __shared__ float  sRaw[2 * NF * DD];  // double-buffered raw F (2 x 32 KB)
#endif

  const int tid  = threadIdx.x;
  const int lane = tid & 31;
  const int wid  = tid >> 5;
  const int b0   = blockIdx.x * GB;
  const int n    = lane & 15;
  const int kbb  = (lane >> 4) << 3;   // B-frag K base: 0/8
  const int kba  = (lane >> 4) << 3;   // A-frag K base: 0/8
  const int mb   = (lane >> 4) << 3;   // C/D M base: 0/8

#if HAVE_ASYNC
  stage_async(F + (size_t)b0 * NF * DD, sRaw, tid);
#endif

  // =========================== Phase A ===========================
  for (int bi = 0; bi < GB; ++bi) {
    const int b = b0 + bi;

#if HAVE_ASYNC
    // Issue next batch's async copy, then wait for the current buffer.
    if (bi + 1 < GB) {
      stage_async(F + (size_t)(b + 1) * NF * DD,
                  sRaw + ((bi + 1) & 1) * NF * DD, tid);
      WAITA(8);       // oldest 8 (current buffer) complete, in-order
    } else {
      WAITA(0);
    }
    __syncthreads();  // all waves' async copies of current buffer visible

    // LDS fp32 -> LDS bf16 with padded rows
    {
      const float* cur = sRaw + (bi & 1) * NF * DD;
      for (int idx = tid; idx < NF * DD / 4; idx += 256) {
        const float4 f = *(const float4*)(cur + idx * 4);
        const int r = (idx << 2) / DD;
        const int c = (idx << 2) % DD;
        __bf16* d = &sF[r * ROWP + c];
        d[0] = (__bf16)f.x; d[1] = (__bf16)f.y;
        d[2] = (__bf16)f.z; d[3] = (__bf16)f.w;
      }
    }
#else
    // Fallback: direct global loads + convert
    const float4* Fb = (const float4*)(F + (size_t)b * NF * DD);
    for (int idx = tid; idx < NF * DD / 4; idx += 256) {
      float4 f = Fb[idx];
      const int r = (idx << 2) / DD;
      const int c = (idx << 2) % DD;
      __bf16* d = &sF[r * ROWP + c];
      d[0] = (__bf16)f.x; d[1] = (__bf16)f.y;
      d[2] = (__bf16)f.z; d[3] = (__bf16)f.w;
    }
    if (bi + 1 < GB)
      __builtin_prefetch((const char*)(F + (size_t)(b + 1) * NF * DD) + tid * 128, 0, 3);
#endif
    __syncthreads();

    // ---- GEMM1 + residual in one pass (shared A fragments):
    //      P   = F_b @ Kw^T  -> sP (bf16)
    //      Out = F_b @ Qw^T  -> global (fp32 stores)
    {
      const int rt = wid >> 1;           // row tile (0..3), both outputs
      const int e0 = (wid & 1) * 4;      // 4 N-tiles per wave
      v8f accP[4] = {};
      v8f accR[4] = {};
#pragma unroll
      for (int k0 = 0; k0 < DD; k0 += 32) {
        v16bf a = ld_a(&sF[(rt * 16) * ROWP + k0], ROWP, lane);
#pragma unroll
        for (int t = 0; t < 4; ++t) {
          const int et = e0 + t;
          v16bf bk = ld_b(Kwbf + (size_t)(et * 16 + n) * DD + k0 + kbb);
          accP[t] = wmma_bf16(a, bk, accP[t]);
          v16bf bq = ld_b(Qwbf + (size_t)(et * 16 + n) * DD + k0 + kbb);
          accR[t] = wmma_bf16(a, bq, accR[t]);
        }
      }
      float* Ob = Out + (size_t)b * NF * DD;
#pragma unroll
      for (int t = 0; t < 4; ++t) {
        const int et = e0 + t;
#pragma unroll
        for (int r = 0; r < 8; ++r) {
          sP[(rt * 16 + mb + r) * ROWP + et * 16 + n] = (__bf16)accP[t][r];
          Ob[(rt * 16 + mb + r) * DD + et * 16 + n]   = accR[t][r];
        }
      }
    }
    __syncthreads();

    // ---- GEMM2: S = F_b @ P^T (M=64,N=64,K=128), B^T storage = sP ----
    {
      const int it = wid >> 1;
      const int j0 = (wid & 1) * 2;
      v8f acc[2] = {};
#pragma unroll
      for (int k0 = 0; k0 < DD; k0 += 32) {
        v16bf a = ld_a(&sF[(it * 16) * ROWP + k0], ROWP, lane);
#pragma unroll
        for (int t = 0; t < 2; ++t) {
          const int jt = j0 + t;
          v16bf bv = ld_b(&sP[(jt * 16 + n) * ROWP + k0 + kbb]);
          acc[t] = wmma_bf16(a, bv, acc[t]);
        }
      }
      __bf16* Sb = &sS[bi * SROW];
#pragma unroll
      for (int t = 0; t < 2; ++t) {
        const int jt = j0 + t;
#pragma unroll
        for (int r = 0; r < 8; ++r)
          Sb[(it * 16 + mb + r) * NF + jt * 16 + n] = (__bf16)acc[t][r];
      }
    }
    __syncthreads();   // protects sF/sP for next iteration; fences sS + Out
  }

  // =========================== Phase B ===========================
  // Wave wid owns d-tile [wid*16, wid*16+15].  For each i:
  //   A[m][j] = sS[m][i*64+j]  (M=16 batches, K=64)
  //   B^T     = Wt[i][d][j]    (N-major, L2-resident)
  const int row = lane & 15;
  for (int i = 0; i < NF; ++i) {
    if (i + 1 < NF)   // prefetch next W[i] slab for this wave's d-tile
      __builtin_prefetch((const char*)(Wt + ((size_t)(i + 1) * DD + wid * 16) * NF)
                             + lane * 128, 0, 3);
    v8f acc = {};
#pragma unroll
    for (int k0 = 0; k0 < NF; k0 += 32) {
      const __bf16* pa = &sS[row * SROW + i * NF + k0 + kba];
      V16U ua;
      ua.h[0] = *(const v8bf*)(pa);
      ua.h[1] = *(const v8bf*)(pa + 16);
      v16bf bv = ld_b(Wt + ((size_t)i * DD + wid * 16 + n) * NF + k0 + kbb);
      acc = wmma_bf16(ua.v, bv, acc);
    }
#pragma unroll
    for (int r = 0; r < 8; ++r) {
      const size_t o = ((size_t)(b0 + mb + r) * NF + i) * (size_t)DD + wid * 16 + n;
      Out[o] += acc[r];   // RMW: line written by this WG in phase A, L2-hot
    }
  }
}

// ---------------------------------------------------------------------------
extern "C" void kernel_launch(void* const* d_in, const int* in_sizes, int n_in,
                              void* d_out, int out_size, void* d_ws, size_t ws_size,
                              hipStream_t stream) {
  (void)in_sizes; (void)n_in; (void)out_size; (void)ws_size;
  const float* F  = (const float*)d_in[0];
  const float* W  = (const float*)d_in[1];
  const float* Kw = (const float*)d_in[2];
  const float* Qw = (const float*)d_in[3];
  float* Out = (float*)d_out;

  char* ws = (char*)d_ws;
  __bf16* Kwbf = (__bf16*)(ws);                   // 32 KB
  __bf16* Qwbf = (__bf16*)(ws + 32768);           // 32 KB
  __bf16* Wt   = (__bf16*)(ws + 65536);           // 1 MB

  k_convert<<<(NF * NF * DD + 255) / 256, 256, 0, stream>>>(Kw, Qw, W, Kwbf, Qwbf, Wt);
  k_fused<<<BB / GB, 256, 0, stream>>>(F, Kwbf, Qwbf, Wt, Out);
}
